// MultiGraphNeuralNetwork_16320875725319
// MI455X (gfx1250) — matrix operand order repaired and checked by reading, not compiled, not run on hardware
//
#include <hip/hip_runtime.h>
#include <cstdint>

// ---------------------------------------------------------------------------
// MultiGraphNeuralNetwork for MI455X (gfx1250, wave32, WMMA bf16)
//
//  1) cvt fp32->bf16: terms1 (14*N^2), terms2 (6*N^2), w2 (2048*448)
//  2) pack x -> x_bf [64 x 2048] and identity rows of z1
//  3) GEMM (wmma bf16 + async-to-LDS B staging): xS1[t] = x_bf @ terms1[t]
//  4) combine1 (VALU fp32): h1 = sigmoid(w1 . z1 + b1)
//  5) GEMM: hS[t] = h1 @ terms2[t]   -> z2[B][7][64][N]
//  6) GEMM: h2[b] = sigmoid(w2_bf [2048x448] @ z2[b] + b2) -> bf16
//  7) linear: out[b,f,:2] = |sigmoid(h2[b,f,:] . w_lin^T + b_lin)|
// ---------------------------------------------------------------------------

#define NN 2048
#define BB 8

#if __has_builtin(__builtin_amdgcn_global_load_async_to_lds_b128) && \
    __has_builtin(__builtin_amdgcn_s_wait_asynccnt)
#define HAS_ASYNC_LDS 1
#else
#define HAS_ASYNC_LDS 0
#endif

typedef __attribute__((ext_vector_type(16))) __bf16 bf16x16;
typedef __attribute__((ext_vector_type(8)))  __bf16 bf16x8;
typedef __attribute__((ext_vector_type(8)))  float  f32x8;
typedef int v4i __attribute__((vector_size(16)));   // matches builtin param type

#if HAS_ASYNC_LDS
typedef __attribute__((address_space(1))) v4i* gvec_ptr;   // global <4 x i32>*
typedef __attribute__((address_space(3))) v4i* lvec_ptr;   // LDS    <4 x i32>*
#endif

__device__ __forceinline__ unsigned short f2bfu(float f) {
    unsigned u = __builtin_bit_cast(unsigned, f);
    unsigned r = u + 0x7fffu + ((u >> 16) & 1u);   // round-to-nearest-even
    return (unsigned short)(r >> 16);
}
__device__ __forceinline__ float bf2f(unsigned short h) {
    unsigned u = ((unsigned)h) << 16;
    return __builtin_bit_cast(float, u);
}
__device__ __forceinline__ float sigmoidf_(float v) {
    return 1.0f / (1.0f + __expf(-v));
}

// --------------------------- fp32 -> bf16 convert ---------------------------
__global__ __launch_bounds__(256)
void cvt_f32_bf16_kernel(const float* __restrict__ src,
                         unsigned short* __restrict__ dst, long long n4) {
    long long i = (long long)blockIdx.x * 256 + threadIdx.x;
    const long long stride = (long long)gridDim.x * 256;
    for (; i < n4; i += stride) {
        float4 v = ((const float4*)src)[i];
        ushort4 o;
        o.x = f2bfu(v.x); o.y = f2bfu(v.y); o.z = f2bfu(v.z); o.w = f2bfu(v.w);
        ((ushort4*)dst)[i] = o;
    }
}

// x [8,8,N] fp32 -> x_bf [64 x N] bf16 (contiguous) + z1 identity rows
__global__ __launch_bounds__(256)
void pack_x_kernel(const float* __restrict__ x,
                   unsigned short* __restrict__ x_bf,
                   unsigned short* __restrict__ z1) {
    int i = blockIdx.x * 256 + threadIdx.x;        // 0 .. 131071
    unsigned short v = f2bfu(x[i]);
    x_bf[i] = v;
    int b   = i >> 14;                             // / (8*2048)
    int rem = i & 16383;                           // g*N + n
    z1[(long long)b * 120 * NN + rem] = v;
}

// ------------------------- generic bf16 WMMA GEMM ---------------------------
// C_z = act(A_z[M x K] * B_z[K x Ncols] (+ bias[row]))  in bf16, fp32 accum.
// Row remap for C: offset(row) = (row>>rpgShift)*groupStride + (row&mask)*ldc.
// Block: 256 thr = 8 waves; tile 64(M) x 256(N); wave tile 32 x 64; K step 32.
// B tile (32 x 256 bf16) double-buffered in LDS via GLOBAL_LOAD_ASYNC_TO_LDS,
// row stride padded to 528B so the 16-bit B-fragment reads of the two lane
// halves (8 rows apart) are staggered across LDS banks.
#define BTILE_LDB 528            // bytes per LDS B-tile row (512 data + 16 pad)
#define BTILE_BYTES (32 * BTILE_LDB)

__global__ __launch_bounds__(256)
void gemm_bf16_wmma(const unsigned short* __restrict__ Ag, long long strideA, int lda,
                    const unsigned short* __restrict__ Bg, long long strideB, int ldb,
                    unsigned short* __restrict__ Cg, long long strideC, int ldc,
                    int K,
                    int rpgShift, long long groupStride,
                    const float* __restrict__ bias, int act) {
    const int lane = threadIdx.x & 31;
    const int wave = threadIdx.x >> 5;
    const int wm = wave >> 2;           // 0..1
    const int wn = wave & 3;            // 0..3
    const long long z = blockIdx.z;

    const __bf16* A = (const __bf16*)(Ag + z * strideA);
    const __bf16* B = (const __bf16*)(Bg + z * strideB);
    unsigned short* C = Cg + z * strideC;

    const int mblk = blockIdx.y * 64 + wm * 32;
    const int nblk = blockIdx.x * 256 + wn * 64;

    f32x8 acc[2][4];
#pragma unroll
    for (int mi = 0; mi < 2; ++mi)
#pragma unroll
        for (int ni = 0; ni < 4; ++ni) acc[mi][ni] = (f32x8)0.0f;

    // A fragment base: lane L<16 holds row M=L, K {0..7,16..23}; L>=16 -> K {8..15,24..31}
    const int arow = mblk + (lane & 15);
    const int aoff = (lane >> 4) * 8;
    const __bf16* pa0 = A + (long long)arow * lda + aoff;
    const __bf16* pa1 = pa0 + 16LL * lda;

#if HAS_ASYNC_LDS
    __shared__ __attribute__((aligned(16))) unsigned char ldsB[2][BTILE_BYTES];
    const unsigned char* Bbytes = (const unsigned char*)B;
    const long long ldbB = (long long)ldb * 2;
    const long long nblkB = (long long)(blockIdx.x * 256) * 2;

    // prologue: async-fill buffer 0 with K-tile 0 (each thread 4 x b128)
    {
        const unsigned char* gb = Bbytes + nblkB;
#pragma unroll
        for (int i = 0; i < 4; ++i) {
            int c = (int)threadIdx.x + i * 256;      // 0..1023 chunks of 16B
            int row = c >> 5, cc = c & 31;
            __builtin_amdgcn_global_load_async_to_lds_b128(
                (gvec_ptr)(gb + (long long)row * ldbB + cc * 16),
                (lvec_ptr)&ldsB[0][row * BTILE_LDB + cc * 16],
                0, 0);
        }
    }
    int buf = 0;
#else
    const __bf16* pb = B + (long long)((lane >> 4) * 8) * ldb + nblk + (lane & 15);
#endif

    for (int k0 = 0; k0 < K; k0 += 32) {
#if HAS_ASYNC_LDS
        __builtin_amdgcn_s_wait_asynccnt(0);   // our slice of current buffer landed
        __syncthreads();                       // everyone's slice landed
        if (k0 + 32 < K) {                     // async-fill the other buffer
            const unsigned char* gb = Bbytes + (long long)(k0 + 32) * ldbB + nblkB;
#pragma unroll
            for (int i = 0; i < 4; ++i) {
                int c = (int)threadIdx.x + i * 256;
                int row = c >> 5, cc = c & 31;
                __builtin_amdgcn_global_load_async_to_lds_b128(
                    (gvec_ptr)(gb + (long long)row * ldbB + cc * 16),
                    (lvec_ptr)&ldsB[buf ^ 1][row * BTILE_LDB + cc * 16],
                    0, 0);
            }
        }
#endif
        // ---- A fragments (direct global, cached) ----
        bf16x8 a0lo = *(const bf16x8*)(pa0 + k0);
        bf16x8 a0hi = *(const bf16x8*)(pa0 + k0 + 16);
        bf16x8 a1lo = *(const bf16x8*)(pa1 + k0);
        bf16x8 a1hi = *(const bf16x8*)(pa1 + k0 + 16);
        bf16x16 afrag[2];
        afrag[0] = __builtin_shufflevector(a0lo, a0hi, 0,1,2,3,4,5,6,7,8,9,10,11,12,13,14,15);
        afrag[1] = __builtin_shufflevector(a1lo, a1hi, 0,1,2,3,4,5,6,7,8,9,10,11,12,13,14,15);
        if (k0 + 32 < K)
            __builtin_prefetch((const void*)(pa0 + k0 + 32), 0, 1);

        // ---- B fragments ----
        bf16x16 bfrag[4];
#if HAS_ASYNC_LDS
        {
            const unsigned char* lb = &ldsB[buf][(unsigned)((lane >> 4) * 8) * BTILE_LDB + (unsigned)(wn * 64 + (lane & 15)) * 2];
#pragma unroll
            for (int ni = 0; ni < 4; ++ni) {
#pragma unroll
                for (int s = 0; s < 16; ++s)
                    bfrag[ni][s] = *(const __bf16*)(lb + ((s & 7) + (s >> 3) * 16) * BTILE_LDB + ni * 32);
            }
        }
#else
#pragma unroll
        for (int ni = 0; ni < 4; ++ni)
#pragma unroll
            for (int s = 0; s < 16; ++s)
                bfrag[ni][s] = pb[(long long)(k0 + (s & 7) + (s >> 3) * 16) * ldb + ni * 16];
#endif

#pragma unroll
        for (int mi = 0; mi < 2; ++mi)
#pragma unroll
            for (int ni = 0; ni < 4; ++ni)
                acc[mi][ni] = __builtin_amdgcn_wmma_f32_16x16x32_bf16(
                    false, afrag[mi], false, bfrag[ni],
                    (short)0, acc[mi][ni], false, false);

#if HAS_ASYNC_LDS
        __syncthreads();                       // done reading buf before overwrite
        buf ^= 1;
#endif
    }

    // Epilogue: C/D layout -> VGPR r: lane<16 row M=r, lane>=16 row M=r+8; col = lane&15
    const int mask = (1 << rpgShift) - 1;
#pragma unroll
    for (int mi = 0; mi < 2; ++mi) {
#pragma unroll
        for (int r = 0; r < 8; ++r) {
            const int gm = mblk + mi * 16 + ((lane < 16) ? r : (r + 8));
            const long long rowOff = (long long)(gm >> rpgShift) * groupStride
                                   + (long long)(gm & mask) * ldc;
            float bv = (act && bias) ? bias[gm] : 0.0f;
#pragma unroll
            for (int ni = 0; ni < 4; ++ni) {
                const int col = nblk + ni * 16 + (lane & 15);
                float v = acc[mi][ni][r];
                if (act) v = sigmoidf_(v + bv);
                C[rowOff + col] = f2bfu(v);
            }
        }
    }
}

// --------------------------- combine-1 (tiny, VALU) -------------------------
__global__ __launch_bounds__(256)
void combine1_kernel(const unsigned short* __restrict__ z1,
                     const float* __restrict__ w1,
                     const float* __restrict__ b1,
                     unsigned short* __restrict__ h1,
                     unsigned short* __restrict__ z2) {
    const int n = blockIdx.x * 256 + threadIdx.x;
    const int f = blockIdx.y;
    const int b = blockIdx.z;
    const unsigned short* zr = z1 + (long long)b * 120 * NN + n;
    const float* wf = w1 + f * 120;     // uniform per block -> scalar loads
    float acc = b1[f];
#pragma unroll 8
    for (int kk = 0; kk < 120; ++kk)
        acc += wf[kk] * bf2f(zr[(long long)kk * NN]);
    unsigned short us = f2bfu(sigmoidf_(acc));
    h1[((long long)b * 64 + f) * NN + n] = us;                 // [512 x N]
    z2[((long long)b * 448 + f) * NN + n] = us;                // z2[b][0][f][n]
}

// ------------------------------ final linear --------------------------------
__global__ __launch_bounds__(256)
void linear_kernel(const unsigned short* __restrict__ h2,
                   const float* __restrict__ wlin,
                   const float* __restrict__ blin,
                   float* __restrict__ out) {
    const int lane = threadIdx.x & 31;
    const int row = blockIdx.x * 8 + (threadIdx.x >> 5);   // 0 .. 16383
    const unsigned short* hr = h2 + (long long)row * NN;
    float a0 = 0.0f, a1 = 0.0f;
    for (int n = lane; n < NN; n += 32) {
        float h = bf2f(hr[n]);
        a0 += h * wlin[n];
        a1 += h * wlin[NN + n];
    }
#pragma unroll
    for (int off = 16; off > 0; off >>= 1) {
        a0 += __shfl_xor(a0, off, 32);
        a1 += __shfl_xor(a1, off, 32);
    }
    if (lane == 0) {
        out[(long long)row * 2 + 0] = fabsf(sigmoidf_(a0 + blin[0]));
        out[(long long)row * 2 + 1] = fabsf(sigmoidf_(a1 + blin[1]));
    }
}

// ------------------------------- launch glue --------------------------------
extern "C" void kernel_launch(void* const* d_in, const int* in_sizes, int n_in,
                              void* d_out, int out_size, void* d_ws, size_t ws_size,
                              hipStream_t stream) {
    const float* x      = (const float*)d_in[0];  // [8,8,2048]
    const float* terms1 = (const float*)d_in[1];  // [14,2048,2048]
    const float* terms2 = (const float*)d_in[2];  // [6,2048,2048]
    const float* w1     = (const float*)d_in[3];  // [64,15,8]
    const float* b1     = (const float*)d_in[4];  // [64,1]
    const float* w2     = (const float*)d_in[5];  // [2048,7,64]
    const float* b2     = (const float*)d_in[6];  // [2048,1]
    const float* wlin   = (const float*)d_in[7];  // [2,2048]
    const float* blin   = (const float*)d_in[8];  // [2]
    float* out = (float*)d_out;

    char* ws = (char*)d_ws;
    size_t off = 0;
    auto carve = [&](size_t bytes) {
        void* p = ws + off;
        off = (off + bytes + 255) & ~(size_t)255;
        return p;
    };
    const long long N2 = (long long)NN * NN;
    unsigned short* t1b  = (unsigned short*)carve(14ull * N2 * 2);        // terms1 bf16
    unsigned short* t2b  = (unsigned short*)carve(6ull  * N2 * 2);        // terms2 bf16
    unsigned short* xbf  = (unsigned short*)carve(64ull * NN * 2);        // x as [64 x N]
    unsigned short* z1   = (unsigned short*)carve((long long)BB*120*NN*2);// [B][15][8][N]
    unsigned short* h1b  = (unsigned short*)carve(512ull * NN * 2);       // [512 x N]
    unsigned short* z2   = (unsigned short*)carve((long long)BB*448*NN*2);// [B][7][64][N]
    unsigned short* w2b  = (unsigned short*)carve(2048ull * 448 * 2);     // [2048 x 448]
    unsigned short* h2   = (unsigned short*)carve((long long)BB*2048*NN*2);// [B][2048][N]

    // 1) conversions
    cvt_f32_bf16_kernel<<<8192, 256, 0, stream>>>(terms1, t1b, 14ll * N2 / 4);
    cvt_f32_bf16_kernel<<<8192, 256, 0, stream>>>(terms2, t2b, 6ll * N2 / 4);
    cvt_f32_bf16_kernel<<<1024, 256, 0, stream>>>(w2,     w2b, 2048ll * 448 / 4);
    // 2) x -> bf16 + z1 identity rows
    pack_x_kernel<<<512, 256, 0, stream>>>(x, xbf, z1);

    // 3) diffusion 1: 14 terms; C scattered into z1 (k=t+1 slots)
    gemm_bf16_wmma<<<dim3(NN / 256, 64 / 64, 14), 256, 0, stream>>>(
        xbf, 0, NN,
        t1b, N2, NN,
        z1 + 8ll * NN, 8ll * NN, NN,
        NN, /*rpgShift=*/3, /*groupStride=*/120ll * NN,
        nullptr, /*act=*/0);

    // 4) combine 1 -> h1 (and z2 k=0 slot)
    combine1_kernel<<<dim3(NN / 256, 64, BB), 256, 0, stream>>>(z1, w1, b1, h1b, z2);

    // 5) diffusion 2: 6 terms; C scattered into z2 (k=t+1 slots)
    gemm_bf16_wmma<<<dim3(NN / 256, 512 / 64, 6), 256, 0, stream>>>(
        h1b, 0, NN,
        t2b, N2, NN,
        z2 + 64ll * NN, 64ll * NN, NN,
        NN, /*rpgShift=*/6, /*groupStride=*/448ll * NN,
        nullptr, /*act=*/0);

    // 6) combine 2: per-batch [2048x448] @ [448x2048] + b2, sigmoid -> h2 bf16
    gemm_bf16_wmma<<<dim3(NN / 256, 2048 / 64, BB), 256, 0, stream>>>(
        w2b, 0, 448,
        z2, 448ll * NN, NN,
        h2, 2048ll * NN, NN,
        448, /*rpgShift=*/11, /*groupStride=*/0,
        b2, /*act=*/1);

    // 7) final linear + sigmoid + abs -> fp32 out [8,2048,2]
    linear_kernel<<<2048, 256, 0, stream>>>(h2, wlin, blin, out);

    (void)in_sizes; (void)n_in; (void)out_size; (void)ws_size;
}
